// Attention_19645180412183
// MI455X (gfx1250) — matrix operand order repaired and checked by reading, not compiled, not run on hardware
//
#include <hip/hip_runtime.h>
#include <hip/hip_bf16.h>

typedef __bf16 bf16;
typedef __attribute__((ext_vector_type(16))) __bf16 v16bf;
typedef __attribute__((ext_vector_type(8)))  float  v8f;

#define DEV __device__ __forceinline__

constexpr int Bq = 8, Pq = 1024, Dq = 1024, Hq = 16, DHq = 64, HIDq = 4096;
constexpr int Mq = Bq * Pq;                 // 8192 token rows
constexpr float SCALEq = 0.125f;            // DH^-0.5

// ---------------------------------------------------------------------------
// WMMA fragment helpers (A/B in this file's contiguous-K order, not the ISA table order, wave32)
// A 16x32 bf16: lane<16 -> K0..15 (row=lane); lane>=16 -> K16..31 (row=lane-16)
// B 32x16 bf16: lane<16 -> K0..15 (col=lane); lane>=16 -> K16..31 (col=lane-16)
// C 16x16 f32 : vgpr r -> row r (lanes 0-15) / row r+8 (lanes 16-31), col=lane%16
// ---------------------------------------------------------------------------
DEV v16bf frag_a(const bf16* p) {           // p -> element (row, kb)
  bf16 t[16];
  *(uint4*)(&t[0]) = *(const uint4*)(p);
  *(uint4*)(&t[8]) = *(const uint4*)(p + 8);
  v16bf r; __builtin_memcpy(&r, t, 32); return r;
}
DEV v16bf frag_b(const bf16* p) {           // p -> element (kb, col): 16 contiguous K
  bf16 t[16];
  *(uint4*)(&t[0]) = *(const uint4*)(p);
  *(uint4*)(&t[8]) = *(const uint4*)(p + 8);
  v16bf r; __builtin_memcpy(&r, t, 32); return r;
}
DEV v8f zero8() {
  v8f z;
#pragma unroll
  for (int i = 0; i < 8; ++i) z[i] = 0.f;
  return z;
}
DEV float gelu_f(float x) { return 0.5f * x * (1.f + erff(x * 0.70710678118654752f)); }

// CDNA5 async global->LDS copy (16B per lane), ASYNCcnt-tracked.
// INST_OFFSET applies to both LDS and global addresses (ISA 08_async_tensor §4.4).
DEV void async_copy_b128x2(unsigned lds_addr, const void* gaddr) {
  unsigned long long ga = (unsigned long long)(uintptr_t)gaddr;
  asm volatile("global_load_async_to_lds_b128 %0, %1, off"
               :: "v"(lds_addr), "v"(ga) : "memory");
  asm volatile("global_load_async_to_lds_b128 %0, %1, off offset:16"
               :: "v"(lds_addr), "v"(ga) : "memory");
}

// ---------------------------------------------------------------------------
// Weight convert + transpose: wT[n][k] = bf16(w[k][n]);  K,N multiples of 32
// ---------------------------------------------------------------------------
__global__ __launch_bounds__(256) void transpose_convert_kernel(
    const float* __restrict__ w, bf16* __restrict__ wT, int K, int N) {
  __shared__ float tile[32][33];
  int tx = threadIdx.x, ty = threadIdx.y;
  int n0 = blockIdx.x * 32, k0 = blockIdx.y * 32;
#pragma unroll
  for (int j = 0; j < 4; ++j)
    tile[ty + j * 8][tx] = w[(size_t)(k0 + ty + j * 8) * N + n0 + tx];
  __syncthreads();
#pragma unroll
  for (int j = 0; j < 4; ++j) {
    int nr = ty + j * 8;
    wT[(size_t)(n0 + nr) * K + k0 + tx] = (bf16)tile[tx][nr];
  }
}

// ---------------------------------------------------------------------------
// LayerNorm over D=1024, one row per block (256 threads), bf16 output
// ---------------------------------------------------------------------------
__global__ __launch_bounds__(256) void layernorm_bf16_kernel(
    const float* __restrict__ x, const float* __restrict__ w,
    const float* __restrict__ b, bf16* __restrict__ out, int fastpass) {
  int m = blockIdx.x;
  int rr = fastpass ? (m + m / Pq) : m;
  const float* xp = x + (size_t)rr * Dq;
  int tid = threadIdx.x;
  float v[4], s = 0.f, ss = 0.f;
#pragma unroll
  for (int i = 0; i < 4; ++i) {
    v[i] = xp[i * 256 + tid];
    s += v[i]; ss += v[i] * v[i];
  }
#pragma unroll
  for (int msk = 1; msk < 32; msk <<= 1) {
    s += __shfl_xor(s, msk, 32);
    ss += __shfl_xor(ss, msk, 32);
  }
  __shared__ float sh[16];
  int wid = tid >> 5, lane = tid & 31;
  if (lane == 0) { sh[wid] = s; sh[8 + wid] = ss; }
  __syncthreads();
  s = 0.f; ss = 0.f;
#pragma unroll
  for (int i = 0; i < 8; ++i) { s += sh[i]; ss += sh[8 + i]; }
  float mu = s * (1.f / Dq);
  float var = ss * (1.f / Dq) - mu * mu;
  float rstd = rsqrtf(var + 1e-5f);
  bf16* op = out + (size_t)m * Dq;
#pragma unroll
  for (int i = 0; i < 4; ++i) {
    int c = i * 256 + tid;
    op[c] = (bf16)((v[i] - mu) * rstd * w[c] + b[c]);
  }
}

// ---------------------------------------------------------------------------
// bf16 WMMA GEMM: C[M,N] = A[M,K] @ W[K,N] (+bias)(GELU?)(+resid)
// WT is W transposed [N,K]. 256 thr, tile 128x128, BK=32, 8 waves of 32x64.
// Double-buffered LDS fed by global_load_async_to_lds_b128 (ASYNCcnt).
// ---------------------------------------------------------------------------
__global__ __launch_bounds__(256) void gemm_bf16_kernel(
    const bf16* __restrict__ A, const bf16* __restrict__ WT,
    const float* __restrict__ bias, const float* __restrict__ resid,
    float* __restrict__ outF, bf16* __restrict__ outB,
    int M, int N, int K, int residFast, int outFast, int doGelu) {
  __shared__ bf16 As[2][128 * 40];
  __shared__ bf16 Bs[2][128 * 40];
  int tid = threadIdx.x, wid = tid >> 5, lane = tid & 31;
  int l15 = lane & 15, half = lane >> 4;
  int wm = (wid & 3) * 32, wn = (wid >> 2) * 64;
  int mBase = blockIdx.y * 128, nBase = blockIdx.x * 128;

  v8f acc[2][4];
#pragma unroll
  for (int i = 0; i < 2; ++i)
#pragma unroll
    for (int j = 0; j < 4; ++j) acc[i][j] = zero8();

  int arow = tid >> 1, koff = (tid & 1) << 4;     // each thread: 32B of one row
  const bf16* aG = A  + (size_t)(mBase + arow) * K + koff;
  const bf16* bG = WT + (size_t)(nBase + arow) * K + koff;
  unsigned aL[2], bL[2];
#pragma unroll
  for (int u = 0; u < 2; ++u) {
    aL[u] = (unsigned)(uintptr_t)&As[u][arow * 40 + koff];
    bL[u] = (unsigned)(uintptr_t)&Bs[u][arow * 40 + koff];
  }

  int steps = K >> 5;
  // prefetch tile 0 (4 async b128 copies per thread)
  async_copy_b128x2(aL[0], aG);
  async_copy_b128x2(bL[0], bG);

  for (int s = 0; s < steps; ++s) {
    int cur = s & 1;
    if (s + 1 < steps) {                 // prefetch next tile into other buffer
      async_copy_b128x2(aL[cur ^ 1], aG + (size_t)(s + 1) * 32);
      async_copy_b128x2(bL[cur ^ 1], bG + (size_t)(s + 1) * 32);
      asm volatile("s_wait_asynccnt 0x4" ::: "memory");  // current tile done
    } else {
      asm volatile("s_wait_asynccnt 0x0" ::: "memory");
    }
    __syncthreads();                     // all lanes' LDS writes visible

    v16bf af[2], bw[4];
#pragma unroll
    for (int i = 0; i < 2; ++i)
      af[i] = frag_a(&As[cur][(wm + i * 16 + l15) * 40 + (half ? 16 : 0)]);
#pragma unroll
    for (int j = 0; j < 4; ++j)
      bw[j] = frag_b(&Bs[cur][(wn + j * 16 + l15) * 40 + (half ? 16 : 0)]);
#pragma unroll
    for (int i = 0; i < 2; ++i)
#pragma unroll
      for (int j = 0; j < 4; ++j)
        acc[i][j] = __builtin_amdgcn_wmma_f32_16x16x32_bf16(
            false, af[i], false, bw[j], (short)0, acc[i][j], false, false);
    __syncthreads();                     // reads done before buffer is reused
  }

#pragma unroll
  for (int i = 0; i < 2; ++i) {
#pragma unroll
    for (int j = 0; j < 4; ++j) {
      int col = nBase + wn + j * 16 + l15;
      float bv = bias ? bias[col] : 0.f;
#pragma unroll
      for (int r = 0; r < 8; ++r) {
        int m = mBase + wm + i * 16 + r + half * 8;
        float v = acc[i][j][r] + bv;
        if (doGelu) v = gelu_f(v);
        if (resid) {
          int rm = residFast ? (m + m / Pq) : m;
          v += resid[(size_t)rm * N + col];
        }
        if (outF) {
          int om = outFast ? (m + m / Pq) : m;
          outF[(size_t)om * N + col] = v;
        }
        if (outB) outB[(size_t)m * N + col] = (bf16)v;
      }
    }
  }
}

// ---------------------------------------------------------------------------
// Flash attention: grid (P/128, B*H). 256 thr = 8 waves; wave owns 16 q rows.
// qkv bf16 [B*P, 3*D] (Q|K|V each D wide, head h at h*64). o bf16 [B*P, D].
// ---------------------------------------------------------------------------
__global__ __launch_bounds__(256) void attention_kernel(
    const bf16* __restrict__ qkv, bf16* __restrict__ o) {
  __shared__ bf16 Vt[64 * 32];        // [dh][key] for current 32-key tile
  __shared__ bf16 Pl[8][16 * 32];     // per-wave P staging [row][key]
  int tid = threadIdx.x, wid = tid >> 5, lane = tid & 31;
  int l15 = lane & 15, half = lane >> 4;
  int bh = blockIdx.y;
  int b = bh >> 4, h = bh & 15;
  int q0 = blockIdx.x * 128 + wid * 16;

  const bf16* qbase = qkv + (size_t)(b * Pq) * 3072 + h * 64;
  const bf16* kbase = qbase + 1024;
  const bf16* vbase = qbase + 2048;

  v16bf aq[2];
#pragma unroll
  for (int c = 0; c < 2; ++c)
    aq[c] = frag_a(qbase + (size_t)(q0 + l15) * 3072 + c * 32 + (half ? 16 : 0));

  v8f ofr[4];
#pragma unroll
  for (int f = 0; f < 4; ++f) ofr[f] = zero8();
  float mprev[8], lsum[8];
#pragma unroll
  for (int r = 0; r < 8; ++r) { mprev[r] = -1e30f; lsum[r] = 0.f; }

  for (int kt = 0; kt < Pq; kt += 32) {
    __syncthreads();
    {   // stage V tile transposed: Vt[dh][key]
      int key = tid >> 3, d0 = (tid & 7) << 3;
      const bf16* vp = vbase + (size_t)(kt + key) * 3072 + d0;
      bf16 tmp[8];
      *(uint4*)tmp = *(const uint4*)vp;
#pragma unroll
      for (int j = 0; j < 8; ++j) Vt[(d0 + j) * 32 + key] = tmp[j];
    }
    __syncthreads();

    // scores S[16q x 32k] = Q(16x64) @ K^T(64x32)
    v8f s[2];
    s[0] = zero8(); s[1] = zero8();
#pragma unroll
    for (int kg = 0; kg < 2; ++kg)
#pragma unroll
      for (int c = 0; c < 2; ++c) {
        v16bf bk = frag_b(kbase + (size_t)(kt + kg * 16 + l15) * 3072 +
                          c * 32 + (half ? 16 : 0));
        s[kg] = __builtin_amdgcn_wmma_f32_16x16x32_bf16(
            false, aq[c], false, bk, (short)0, s[kg], false, false);
      }

    // online softmax (per row: 16-lane butterfly within half-wave)
#pragma unroll
    for (int r = 0; r < 8; ++r) {
      float v0 = s[0][r] * SCALEq, v1 = s[1][r] * SCALEq;
      float t = fmaxf(v0, v1);
#pragma unroll
      for (int msk = 1; msk <= 8; msk <<= 1) t = fmaxf(t, __shfl_xor(t, msk, 32));
      float mnew = fmaxf(mprev[r], t);
      float alpha = __expf(mprev[r] - mnew);
      float p0 = __expf(v0 - mnew), p1 = __expf(v1 - mnew);
      float rs = p0 + p1;
#pragma unroll
      for (int msk = 1; msk <= 8; msk <<= 1) rs += __shfl_xor(rs, msk, 32);
      lsum[r] = lsum[r] * alpha + rs;
      mprev[r] = mnew;
#pragma unroll
      for (int f = 0; f < 4; ++f) ofr[f][r] *= alpha;
      int prow = r + half * 8;
      Pl[wid][prow * 32 + l15] = (bf16)p0;
      Pl[wid][prow * 32 + 16 + l15] = (bf16)p1;
    }

    // O += P(16x32) @ V(32x64)
    v16bf ap = frag_a(&Pl[wid][l15 * 32 + (half ? 16 : 0)]);
#pragma unroll
    for (int f = 0; f < 4; ++f) {
      v16bf bv = frag_b(&Vt[(f * 16 + l15) * 32 + (half ? 16 : 0)]);
      ofr[f] = __builtin_amdgcn_wmma_f32_16x16x32_bf16(
          false, ap, false, bv, (short)0, ofr[f], false, false);
    }
  }

  // epilogue: o[b, q, h*64 + dh] = O / lsum
#pragma unroll
  for (int f = 0; f < 4; ++f)
#pragma unroll
    for (int r = 0; r < 8; ++r) {
      int qr = q0 + r + half * 8;
      float val = ofr[f][r] / lsum[r];
      o[(size_t)(b * Pq + qr) * Dq + h * 64 + f * 16 + l15] = (bf16)val;
    }
}

// ---------------------------------------------------------------------------
// finalize: zero fastpass rows of output, copy predictions placeholder
// ---------------------------------------------------------------------------
__global__ __launch_bounds__(256) void finalize_kernel(
    float* __restrict__ out, const float* __restrict__ pred) {
  int idx = blockIdx.x * 256 + threadIdx.x;
  if (idx < Bq * Dq) {
    int b = idx / Dq, d = idx % Dq;
    out[((size_t)b * (Pq + 1) + Pq) * Dq + d] = 0.f;
  } else {
    int j = idx - Bq * Dq;
    if (j < Bq * 1000) out[(size_t)Bq * (Pq + 1) * Dq + j] = pred[j];
  }
}

// ---------------------------------------------------------------------------
extern "C" void kernel_launch(void* const* d_in, const int* in_sizes, int n_in,
                              void* d_out, int out_size, void* d_ws, size_t ws_size,
                              hipStream_t stream) {
  (void)in_sizes; (void)n_in; (void)out_size; (void)ws_size;
  const float* x    = (const float*)d_in[0];
  const float* pred = (const float*)d_in[1];
  const float* ln1w = (const float*)d_in[2];
  const float* ln1b = (const float*)d_in[3];
  const float* wqkv = (const float*)d_in[4];
  const float* bqkv = (const float*)d_in[5];
  const float* wo   = (const float*)d_in[6];
  const float* bo   = (const float*)d_in[7];
  const float* ln2w = (const float*)d_in[8];
  const float* ln2b = (const float*)d_in[9];
  const float* w1   = (const float*)d_in[10];
  const float* b1   = (const float*)d_in[11];
  const float* w2   = (const float*)d_in[12];
  const float* b2   = (const float*)d_in[13];
  float* out = (float*)d_out;

  char* ws = (char*)d_ws;                       // arena layout (bytes)
  bf16*  wqkvT = (bf16*)(ws + 0);               //  6,291,456
  bf16*  woT   = (bf16*)(ws + 6291456);         //  2,097,152
  bf16*  w1T   = (bf16*)(ws + 8388608);         //  8,388,608
  bf16*  w2T   = (bf16*)(ws + 16777216);        //  8,388,608
  bf16*  xn    = (bf16*)(ws + 25165824);        // 16,777,216 (ln1 then ln2)
  float* x2    = (float*)(ws + 41943040);       // 33,554,432
  bf16*  obf   = (bf16*)(ws + 75497472);        // 16,777,216
  bf16*  arena = (bf16*)(ws + 92274688);        // 67,108,864 (qkv then h)

  dim3 t32x8(32, 8);
  transpose_convert_kernel<<<dim3(3072 / 32, 1024 / 32), t32x8, 0, stream>>>(wqkv, wqkvT, 1024, 3072);
  transpose_convert_kernel<<<dim3(1024 / 32, 1024 / 32), t32x8, 0, stream>>>(wo, woT, 1024, 1024);
  transpose_convert_kernel<<<dim3(4096 / 32, 1024 / 32), t32x8, 0, stream>>>(w1, w1T, 1024, 4096);
  transpose_convert_kernel<<<dim3(1024 / 32, 4096 / 32), t32x8, 0, stream>>>(w2, w2T, 4096, 1024);

  layernorm_bf16_kernel<<<Mq, 256, 0, stream>>>(x, ln1w, ln1b, xn, 1);

  // qkv = ln1(x) @ w_qkv + b_qkv  -> bf16 arena
  gemm_bf16_kernel<<<dim3(3072 / 128, Mq / 128), 256, 0, stream>>>(
      xn, wqkvT, bqkv, nullptr, nullptr, arena, Mq, 3072, 1024, 0, 0, 0);

  attention_kernel<<<dim3(Pq / 128, Bq * Hq), 256, 0, stream>>>(arena, obf);

  // x2 = attn_out @ w_o + b_o + x
  gemm_bf16_kernel<<<dim3(1024 / 128, Mq / 128), 256, 0, stream>>>(
      obf, woT, bo, x, x2, nullptr, Mq, 1024, 1024, 1, 0, 0);

  layernorm_bf16_kernel<<<Mq, 256, 0, stream>>>(x2, ln2w, ln2b, xn, 0);

  // h = gelu(ln2(x2) @ w1 + b1) -> bf16 arena
  gemm_bf16_kernel<<<dim3(4096 / 128, Mq / 128), 256, 0, stream>>>(
      xn, w1T, b1, nullptr, nullptr, arena, Mq, 4096, 1024, 0, 0, 1);

  // out = h @ w2 + b2 + x2  (fastpass row mapping into d_out)
  gemm_bf16_kernel<<<dim3(1024 / 128, Mq / 128), 256, 0, stream>>>(
      arena, w2T, b2, x2, out, nullptr, Mq, 1024, 4096, 0, 1, 0);

  finalize_kernel<<<64, 256, 0, stream>>>(out, pred);
}
